// DeepSeekMoE_26877905338905
// MI455X (gfx1250) — matrix operand order repaired and checked by reading, not compiled, not run on hardware
//
#include <hip/hip_runtime.h>
#include <hip/hip_bf16.h>
#include <math.h>

// ---------------- problem constants ----------------
constexpr int kH = 1024;   // hidden
constexpr int kI = 4096;   // intermediate
constexpr int kE = 8;      // experts
constexpr int kT = 2048;   // tokens (B*S)

// ---------------- types ----------------
typedef __attribute__((ext_vector_type(16))) __bf16 v16bf;
typedef __attribute__((ext_vector_type(8)))  float  v8f;
typedef __attribute__((ext_vector_type(4)))  unsigned int u32x4;
typedef __attribute__((ext_vector_type(4)))  unsigned int uint32x4;
typedef __attribute__((ext_vector_type(8)))  int int32x8;
typedef __attribute__((ext_vector_type(4)))  int int32x4;

struct Pair128 { u32x4 lo, hi; };

__device__ __forceinline__ unsigned short f2bf(float f) {
    unsigned int u = __builtin_bit_cast(unsigned int, f);
    unsigned int r = u + 0x7FFFu + ((u >> 16) & 1u);   // round-to-nearest-even
    return (unsigned short)(r >> 16);
}

// load a 32-byte WMMA fragment (16 bf16) from one contiguous LDS/global span
__device__ __forceinline__ v16bf load_frag(const unsigned short* p) {
    const u32x4* q = reinterpret_cast<const u32x4*>(p);
    Pair128 t; t.lo = q[0]; t.hi = q[1];
    return __builtin_bit_cast(v16bf, t);
}
// fragment from two separately addressed b128 halves (pad-aware X tile)
__device__ __forceinline__ v16bf load_frag2(const unsigned short* p0, const unsigned short* p1) {
    Pair128 t;
    t.lo = *reinterpret_cast<const u32x4*>(p0);
    t.hi = *reinterpret_cast<const u32x4*>(p1);
    return __builtin_bit_cast(v16bf, t);
}

// ---------------- kernel 0: zero counters ----------------
__global__ void init_counts_kernel(int* counts) {
    if (threadIdx.x < kE) counts[threadIdx.x] = 0;
}

// ---------------- kernel 1: pack f32 [Emats,K,N] row-major -> bf16 WMMA-B tiles
// tile = 32(K) x 16(N); per lane l: N = nt*16 + (l&15), K = kt*32 + (j>>3)*16 + (l>>4)*8 + (j&7)
__global__ __launch_bounds__(256) void pack_b_kernel(const float* __restrict__ src,
                                                     unsigned short* __restrict__ dst,
                                                     int N, long KN, long total) {
    int ntiles = N >> 4;
    for (long idx = (long)blockIdx.x * 256 + threadIdx.x; idx < total;
         idx += (long)gridDim.x * 256) {
        long e   = idx / KN;
        long rem = idx - e * KN;
        int tile   = (int)(rem >> 9);
        int within = (int)(rem & 511);
        int l = within >> 4, j = within & 15;
        int kt = tile / ntiles, nt = tile - kt * ntiles;
        int Kc = kt * 32 + ((j >> 3) << 4) + ((l >> 4) << 3) + (j & 7);   // B layout: lanes 0-15 K=0..7,16..23; 16-31 K=8..15,24..31
        int Nc = (nt << 4) + (l & 15);
        dst[idx] = f2bf(src[e * KN + (long)Kc * N + Nc]);
    }
}

// ---------------- kernel 2: gating (sigmoid+bias, top-2, normalize) + x->bf16 + routing lists
__global__ __launch_bounds__(256) void gate_kernel(const float* __restrict__ x,
                                                   const float* __restrict__ gw,
                                                   const float* __restrict__ gbias,
                                                   unsigned short* __restrict__ xbf,
                                                   float* __restrict__ topW,
                                                   int* __restrict__ lists,
                                                   int* __restrict__ counts) {
    __shared__ float red[kE * 256];
    int t = blockIdx.x, tid = threadIdx.x;
    const float* xr = x + (size_t)t * kH;
    float acc[kE] = {};
    for (int h = tid; h < kH; h += 256) {
        float xv = xr[h];
        xbf[(size_t)t * kH + h] = f2bf(xv);
        const float* g = gw + (size_t)h * kE;
        #pragma unroll
        for (int e = 0; e < kE; ++e) acc[e] += xv * g[e];
    }
    #pragma unroll
    for (int e = 0; e < kE; ++e) red[e * 256 + tid] = acc[e];
    __syncthreads();
    for (int s = 128; s > 0; s >>= 1) {
        if (tid < s) {
            #pragma unroll
            for (int e = 0; e < kE; ++e) red[e * 256 + tid] += red[e * 256 + tid + s];
        }
        __syncthreads();
    }
    if (tid == 0) {
        float s[kE];
        #pragma unroll
        for (int e = 0; e < kE; ++e) {
            float z = red[e * 256] + gbias[e];
            s[e] = 1.0f / (1.0f + expf(-z));
        }
        int i0 = 0;
        for (int e = 1; e < kE; ++e) if (s[e] > s[i0]) i0 = e;
        int i1 = (i0 == 0) ? 1 : 0;
        for (int e = 0; e < kE; ++e) if (e != i0 && s[e] > s[i1]) i1 = e;
        float inv = 1.0f / (s[i0] + s[i1] + 1e-6f);
        topW[t * 2 + 0] = s[i0] * inv;
        topW[t * 2 + 1] = s[i1] * inv;
        int p0 = atomicAdd(&counts[i0], 1); lists[i0 * kT + p0] = t * 2 + 0;
        int p1 = atomicAdd(&counts[i1], 1); lists[i1 * kT + p1] = t * 2 + 1;
    }
}

// ---------------- kernel 3: fused expert FFN (up->GeLU->down) with WMMA bf16
// grid: x = row tile (16 tokens), y = 0..7 routed experts, 8 = shared expert.
// block = 256 threads = 8 waves; block computes 16 rows x 1024 output cols.
// X tile is gathered into LDS by the Tensor Data Mover (gather mode, 16-bit
// indices, 16B pad per 1024B so ds_load_b128 stays 16B-aligned, rows skewed
// by 8 LDS banks). Row pitch = 2048 + 2*16 = 2080 bytes.
__global__ __launch_bounds__(256) void ffn_kernel(const unsigned short* __restrict__ xbf,
                                                  const unsigned short* __restrict__ pUp,
                                                  const unsigned short* __restrict__ pDown,
                                                  const unsigned short* __restrict__ pSwUp,
                                                  const unsigned short* __restrict__ pSwDown,
                                                  const float* __restrict__ up_b,
                                                  const float* __restrict__ down_b,
                                                  const float* __restrict__ sb_up,
                                                  const float* __restrict__ sb_down,
                                                  const int* __restrict__ lists,
                                                  const int* __restrict__ counts,
                                                  float* __restrict__ eo,
                                                  float* __restrict__ out) {
    __shared__ __attribute__((aligned(16))) unsigned short lds_x[16 * 1040];  // 16 rows * 2080B
    __shared__ __attribute__((aligned(16))) unsigned short lds_act[4 * 512];  // 16x128 act chunk, A layout
    __shared__ int lds_tok[16];

    const int z = blockIdx.y;               // 0..7 expert, 8 shared
    const bool shared_path = (z == kE);
    const int n = shared_path ? kT : counts[z];
    const int rbase = blockIdx.x * 16;
    if (rbase >= n) return;

    const int tid = threadIdx.x, lane = tid & 31, wave = tid >> 5;

    if (tid < 16) {
        int r = rbase + tid;
        int rc = (r < n - 1) ? r : (n - 1);
        lds_tok[tid] = shared_path ? (r << 1) : lists[z * kT + rc];
    }
    __syncthreads();

    // ---- TDM gather: 16 token rows of xbf -> LDS (async tensor path) ----
    if (wave == 0) {
        unsigned ldsOff = (unsigned)(size_t)(void*)lds_x;            // flat LDS addr: offset in [31:0]
        unsigned long long ga = (unsigned long long)(size_t)xbf;     // tensor/tile base
        uint32x4 g0;
        g0[0] = 1u | (0u << 30) | (1u << 31);        // count=1, 16-bit indices, gather_mode=1
        g0[1] = ldsOff;                              // lds_addr
        g0[2] = (unsigned)(ga & 0xFFFFFFFFu);        // global_addr[31:0]
        g0[3] = (unsigned)((ga >> 32) & 0x01FFFFFFu) | (2u << 30);   // addr[56:32], type=2
        int32x8 g1;
        g1[0] = (1 << 16) | (1 << 20) | (7 << 22) | (3 << 25); // data_size=2B, pad_en, interval=256dw, amount=4dw
        g1[1] = (kH & 0xFFFF) << 16;                           // tensor_dim0 lo16 -> bits[63:48]
        g1[2] = ((kH >> 16) & 0xFFFF) | ((kT & 0xFFFF) << 16); // dim0 hi | tensor_dim1 lo
        g1[3] = ((kT >> 16) & 0xFFFF) | (kH << 16);            // dim1 hi | tile_dim0 = 1024
        g1[4] = 16;                                            // tile_dim1 = #valid indices
        g1[5] = kH;                                            // tensor_dim0_stride lo32
        g1[6] = 0;
        g1[7] = 0;
        int32x4 gi0, gi1;
        #pragma unroll
        for (int q = 0; q < 4; ++q) {
            int a0 = lds_tok[2 * q + 0] >> 1, a1 = lds_tok[2 * q + 1] >> 1;
            int b0 = lds_tok[8 + 2 * q + 0] >> 1, b1 = lds_tok[8 + 2 * q + 1] >> 1;
            gi0[q] = (a0 & 0xFFFF) | (a1 << 16);
            gi1[q] = (b0 & 0xFFFF) | (b1 << 16);
        }
#if defined(__clang_major__) && __clang_major__ >= 23
        int32x8 gex = {};
        __builtin_amdgcn_tensor_load_to_lds(g0, g1, gi0, gi1, gex, 0);
#else
        __builtin_amdgcn_tensor_load_to_lds(g0, g1, gi0, gi1, 0);
#endif
        __builtin_amdgcn_s_wait_tensorcnt(0);
    }
    __syncthreads();

    const unsigned short* Bup = shared_path ? pSwUp  : pUp   + (size_t)z * kH * kI;
    const unsigned short* Bdn = shared_path ? pSwDown: pDown + (size_t)z * kI * kH;
    const float* bup = shared_path ? sb_up   : up_b   + z * kI;
    const float* bdn = shared_path ? sb_down : down_b + z * kH;

    v8f acc[8];
    #pragma unroll
    for (int i = 0; i < 8; ++i) acc[i] = v8f{};

    const int colLocal = (wave << 4) + (lane & 15);     // 0..127 within I-chunk (stage 1 cols)
    const int kstep2 = colLocal >> 5;
    const int K32 = colLocal & 31;
    const int lphi = ((K32 >> 3) & 1) << 4;
    const int jact = (((K32 >> 4) & 1) << 3) | (K32 & 7);
    const int rowHalf = (lane >> 4) << 3;

    // stage-1 A fragment base (pad-aware row-major X tile)
    const int rowA = lane & 15;
    const int kbByte = ((lane >> 4) << 3) * 2;          // 0 or 16 bytes
    const char* ldsXc = (const char*)lds_x;

    for (int ch = 0; ch < kI / 128; ++ch) {
        // ---- stage 1: up-projection, wave owns one 16-col n-tile of the 128-chunk ----
        v8f up = v8f{};
        const int nt1 = ch * 8 + wave;                  // I/16 tile index
        for (int kt = 0; kt < kH / 32; ++kt) {
            int d0 = rowA * 2048 + kt * 64 + kbByte;    // linear byte offset in tile
            int d1 = d0 + 32;
            const unsigned short* pa0 = (const unsigned short*)(ldsXc + d0 + ((d0 >> 10) << 4));
            const unsigned short* pa1 = (const unsigned short*)(ldsXc + d1 + ((d1 >> 10) << 4));
            v16bf a = load_frag2(pa0, pa1);
            v16bf b = load_frag(Bup + (((kt << 8) + nt1) << 9) + lane * 16);
            up = __builtin_amdgcn_wmma_f32_16x16x32_bf16(false, a, false, b,
                                                         (short)0, up, false, false);
        }
        __syncthreads();   // prior stage-2 readers done with lds_act
        // bias + exact GeLU, write act chunk into LDS (A-fragment layout)
        {
            float bias = bup[(ch << 7) + colLocal];
            #pragma unroll
            for (int i = 0; i < 8; ++i) {
                int row = rowHalf + i;
                float v = up[i] + bias;
                float g = 0.5f * v * (1.0f + erff(v * 0.70710678118654752f));
                lds_act[(kstep2 << 9) + ((lphi | row) << 4) + jact] = f2bf(g);
            }
        }
        __syncthreads();
        // ---- stage 2: down-projection, wave owns 128 output cols (8 n-tiles) ----
        for (int ks = 0; ks < 4; ++ks) {
            v16bf a = load_frag(lds_act + (ks << 9) + lane * 16);
            int kt2 = (ch << 2) + ks;                   // I/32 tile index
            #pragma unroll
            for (int nn = 0; nn < 8; ++nn) {
                int nt = (wave << 3) + nn;              // H/16 tile index
                v16bf b = load_frag(Bdn + (((kt2 << 6) + nt) << 9) + lane * 16);
                acc[nn] = __builtin_amdgcn_wmma_f32_16x16x32_bf16(false, a, false, b,
                                                                  (short)0, acc[nn], false, false);
            }
        }
        __syncthreads();
    }

    // ---- writeout: routed -> eo[slot], shared -> out[token] (bias added, no 0.1 yet) ----
    #pragma unroll
    for (int nn = 0; nn < 8; ++nn) {
        int col = (wave << 7) + (nn << 4) + (lane & 15);
        float bv = bdn[col];
        #pragma unroll
        for (int i = 0; i < 8; ++i) {
            int row = rowHalf + i;
            if (rbase + row < n) {
                float v = acc[nn][i] + bv;
                if (shared_path) out[(size_t)(rbase + row) * kH + col] = v;
                else             eo[(size_t)lds_tok[row] * kH + col]   = v;
            }
        }
    }
}

// ---------------- kernel 4: combine routed + shared, row-wise abs-max normalize ----------------
__global__ __launch_bounds__(256) void combine_kernel(const float* __restrict__ eo,
                                                      const float* __restrict__ topW,
                                                      float* __restrict__ out) {
    __shared__ float red[256];
    int t = blockIdx.x, tid = threadIdx.x;
    float w0 = topW[t * 2 + 0], w1 = topW[t * 2 + 1];
    const float* e0 = eo + (size_t)(t * 2 + 0) * kH;
    const float* e1 = eo + (size_t)(t * 2 + 1) * kH;
    float* o = out + (size_t)t * kH;
    float v[4]; float mx = 0.0f;
    #pragma unroll
    for (int i = 0; i < 4; ++i) {
        int h = tid + i * 256;
        float val = 0.1f * (o[h] + w0 * e0[h] + w1 * e1[h]);
        v[i] = val;
        mx = fmaxf(mx, fabsf(val));
    }
    red[tid] = mx; __syncthreads();
    for (int s = 128; s > 0; s >>= 1) {
        if (tid < s) red[tid] = fmaxf(red[tid], red[tid + s]);
        __syncthreads();
    }
    float inv = 1.0f / (red[0] + 1e-6f);
    #pragma unroll
    for (int i = 0; i < 4; ++i) o[tid + i * 256] = v[i] * inv;
}

// ---------------- host launcher ----------------
extern "C" void kernel_launch(void* const* d_in, const int* in_sizes, int n_in,
                              void* d_out, int out_size, void* d_ws, size_t ws_size,
                              hipStream_t stream) {
    const float* x       = (const float*)d_in[0];
    const float* gate_w  = (const float*)d_in[1];
    const float* gbias   = (const float*)d_in[2];
    const float* up_w    = (const float*)d_in[3];
    const float* up_b    = (const float*)d_in[4];
    const float* down_w  = (const float*)d_in[5];
    const float* down_b  = (const float*)d_in[6];
    const float* sw_up   = (const float*)d_in[7];
    const float* sb_up   = (const float*)d_in[8];
    const float* sw_down = (const float*)d_in[9];
    const float* sb_down = (const float*)d_in[10];
    float* out = (float*)d_out;

    char* ws = (char*)d_ws;
    size_t off = 0;
    auto carve = [&](size_t bytes) -> void* {
        void* p = ws + off;
        off = (off + bytes + 255) & ~(size_t)255;
        return p;
    };
    unsigned short* xbf     = (unsigned short*)carve((size_t)kT * kH * 2);
    unsigned short* pUp     = (unsigned short*)carve((size_t)kE * kH * kI * 2);
    unsigned short* pDown   = (unsigned short*)carve((size_t)kE * kI * kH * 2);
    unsigned short* pSwUp   = (unsigned short*)carve((size_t)kH * kI * 2);
    unsigned short* pSwDown = (unsigned short*)carve((size_t)kI * kH * 2);
    float*          eo      = (float*)carve((size_t)kT * 2 * kH * 4);
    float*          topW    = (float*)carve((size_t)kT * 2 * 4);
    int*            lists   = (int*)carve((size_t)kE * kT * 4);
    int*            counts  = (int*)carve(64);

    init_counts_kernel<<<1, 32, 0, stream>>>(counts);

    long totE = (long)kE * kH * kI;
    long tot1 = (long)kH * kI;
    pack_b_kernel<<<8192, 256, 0, stream>>>(up_w,   pUp,     kI, (long)kH * kI, totE);
    pack_b_kernel<<<8192, 256, 0, stream>>>(down_w, pDown,   kH, (long)kI * kH, totE);
    pack_b_kernel<<<4096, 256, 0, stream>>>(sw_up,  pSwUp,   kI, (long)kH * kI, tot1);
    pack_b_kernel<<<4096, 256, 0, stream>>>(sw_down,pSwDown, kH, (long)kI * kH, tot1);

    gate_kernel<<<kT, 256, 0, stream>>>(x, gate_w, gbias, xbf, topW, lists, counts);

    ffn_kernel<<<dim3(kT / 16, kE + 1), 256, 0, stream>>>(xbf, pUp, pDown, pSwUp, pSwDown,
                                                          up_b, down_b, sb_up, sb_down,
                                                          lists, counts, eo, out);

    combine_kernel<<<kT, 256, 0, stream>>>(eo, topW, out);
}
